// QKVAttention_59322088292556
// MI455X (gfx1250) — matrix operand order repaired and checked by reading, not compiled, not run on hardware
//
#include <hip/hip_runtime.h>
#include <stdint.h>

// ---------------------------------------------------------------------------
// QKV single-head attention + GELU MLP block, bf16 WMMA pipeline for gfx1250.
// B=8, S=2048, H=1024.  All matmuls via v_wmma_f32_16x16x32_bf16 in "nt" form
// (B operand stored [N][K], k-contiguous): weights are pre-transposed once,
// v is produced transposed by the v-projection epilogue.
// Global->LDS staging uses GLOBAL_LOAD_ASYNC_TO_LDS_B128 (ASYNCcnt) when the
// builtin exists, else a register-prefetch pipeline; 3-slot LDS ring either way.
// ---------------------------------------------------------------------------

typedef __attribute__((ext_vector_type(16))) __bf16 v16bf;
typedef __attribute__((ext_vector_type(8)))  float  v8f;
typedef __attribute__((ext_vector_type(4)))  int    v4i;

static constexpr long long Bc = 8, Sc = 2048, Hc = 1024;
static constexpr long long BS = Bc * Sc;  // 16384 rows

#if __has_builtin(__builtin_amdgcn_global_load_async_to_lds_b128)
#define HAVE_ASYNC_LDS 1
#else
#define HAVE_ASYNC_LDS 0
#endif

typedef __attribute__((address_space(1))) v4i* gv4p;   // global int4*
typedef __attribute__((address_space(3))) v4i* lv4p;   // LDS int4*

__device__ __forceinline__ void async_cp16(const void* g, void* l) {
#if HAVE_ASYNC_LDS
  __builtin_amdgcn_global_load_async_to_lds_b128(
      (gv4p)(uintptr_t)g, (lv4p)(uint32_t)(uintptr_t)l, 0, 0);
#else
  (void)g; (void)l;
#endif
}

template<int N>
__device__ __forceinline__ void wait_asynccnt() {
#if __has_builtin(__builtin_amdgcn_s_wait_asynccnt)
  __builtin_amdgcn_s_wait_asynccnt(N);
#elif HAVE_ASYNC_LDS
  asm volatile("s_wait_asynccnt %0" :: "i"(N) : "memory");
#endif
}

__device__ __forceinline__ uint16_t f2bf(float f) {
  union { float f; uint32_t u; } v; v.f = f;
  uint32_t r = v.u + 0x7FFFu + ((v.u >> 16) & 1u);   // round-to-nearest-even
  return (uint16_t)(r >> 16);
}

__global__ void cvt_f32_bf16(const float* __restrict__ x,
                             uint16_t* __restrict__ y, long long n) {
  long long i  = (long long)blockIdx.x * blockDim.x + threadIdx.x;
  long long st = (long long)gridDim.x * blockDim.x;
  for (; i < n; i += st) y[i] = f2bf(x[i]);
}

// W f32 [K][N] row-major  ->  Wt bf16 [N][K] row-major.  32x32 LDS tiles.
__global__ __launch_bounds__(256) void cvt_transpose_f32_bf16(
    const float* __restrict__ W, uint16_t* __restrict__ Wt, int K, int N) {
  __shared__ float tile[32][33];
  const int n0 = blockIdx.x * 32, k0 = blockIdx.y * 32;
  const int tx = threadIdx.x & 31, ty = threadIdx.x >> 5;  // 32 x 8
#pragma unroll
  for (int i = 0; i < 32; i += 8)
    tile[ty + i][tx] = W[(long long)(k0 + ty + i) * N + n0 + tx];
  __syncthreads();
#pragma unroll
  for (int i = 0; i < 32; i += 8)
    Wt[(long long)(n0 + ty + i) * K + k0 + tx] = f2bf(tile[tx][ty + i]);
}

struct FragBF { union { v16bf v; float4 f4[2]; }; };

// C = A[M,K] * Bt[N,K]^T + bias.  Block tile 128x128x32, 8 waves (4M x 2N),
// each wave 2x4 WMMA 16x16 accumulators.  3-slot LDS ring, pipelined staging.
template<bool DO_GELU, bool OUT_F32, bool OUT_TRANS>
__global__ __launch_bounds__(256) void gemm_nt_bf16_wmma(
    const uint16_t* __restrict__ A, const uint16_t* __restrict__ Bt,
    const float* __restrict__ bias, void* __restrict__ Cout,
    int K, int lda, int ldb, int ldc,
    long long sA, long long sB, long long sC)
{
  A  += (long long)blockIdx.z * sA;
  Bt += (long long)blockIdx.z * sB;

  // Row stride 40 halves = 80B (16B-aligned rows, bank skew).
  __shared__ alignas(16) uint16_t As[3][128][40];
  __shared__ alignas(16) uint16_t Bs[3][128][40];

  const int tid  = threadIdx.x;
  const int wave = tid >> 5, lane = tid & 31;
  const int wm   = wave & 3,  wn  = wave >> 2;
  const int half = lane >> 4, l16 = lane & 15;
  const long long m0 = (long long)blockIdx.y * 128;
  const long long n0 = (long long)blockIdx.x * 128;

  // Per-thread staging: two 16B chunks per matrix per k-step.
  const int r0 = tid >> 2, q0 = (tid & 3) * 8;          // rows 0..63 / 64..127
  const uint16_t* gA0 = A  + (m0 + r0)      * lda + q0;
  const uint16_t* gA1 = A  + (m0 + r0 + 64) * lda + q0;
  const uint16_t* gB0 = Bt + (n0 + r0)      * ldb + q0;
  const uint16_t* gB1 = Bt + (n0 + r0 + 64) * ldb + q0;

  v8f acc[2][4];
#pragma unroll
  for (int i = 0; i < 2; ++i)
#pragma unroll
    for (int j = 0; j < 4; ++j)
#pragma unroll
      for (int e = 0; e < 8; ++e) acc[i][j][e] = 0.0f;

  auto compute = [&](int s) {
    // A (16x32): lanes0-15 row M=l16 hold K{0-7,16-23}; lanes16-31 K{8-15,24-31}
    // B (32x16): lanes0-15 col N=l16 hold K{0-7,16-23}; lanes16-31 K{8-15,24-31}
    FragBF af[2], bfr[4];
#pragma unroll
    for (int i = 0; i < 2; ++i) {
      const float4* r = reinterpret_cast<const float4*>(
          &As[s][wm * 32 + i * 16 + l16][0]);
      af[i].f4[0] = r[half];
      af[i].f4[1] = r[2 + half];
    }
#pragma unroll
    for (int j = 0; j < 4; ++j) {
      const float4* r = reinterpret_cast<const float4*>(
          &Bs[s][wn * 64 + j * 16 + l16][0]);
      bfr[j].f4[0] = r[half];
      bfr[j].f4[1] = r[2 + half];
    }
#pragma unroll
    for (int i = 0; i < 2; ++i)
#pragma unroll
      for (int j = 0; j < 4; ++j)
        acc[i][j] = __builtin_amdgcn_wmma_f32_16x16x32_bf16(
            false, af[i].v, false, bfr[j].v, (short)0, acc[i][j], false, false);
  };

  const int nk = K >> 5;

#if HAVE_ASYNC_LDS
  auto stage = [&](int s, int kk) {
    async_cp16(gA0 + kk, &As[s][r0][q0]);
    async_cp16(gA1 + kk, &As[s][r0 + 64][q0]);
    async_cp16(gB0 + kk, &Bs[s][r0][q0]);
    async_cp16(gB1 + kk, &Bs[s][r0 + 64][q0]);
  };
  stage(0, 0);
  for (int i = 0; i < nk; ++i) {
    const bool next = (i + 1) < nk;
    if (next) stage((i + 1) % 3, (i + 1) * 32);
    if (next) wait_asynccnt<4>(); else wait_asynccnt<0>();
    __syncthreads();
    compute(i % 3);
  }
#else
  float4 pa0, pa1, pb0, pb1;
  auto ldreg = [&](int kk) {
    pa0 = *reinterpret_cast<const float4*>(gA0 + kk);
    pa1 = *reinterpret_cast<const float4*>(gA1 + kk);
    pb0 = *reinterpret_cast<const float4*>(gB0 + kk);
    pb1 = *reinterpret_cast<const float4*>(gB1 + kk);
  };
  auto stor = [&](int s) {
    *reinterpret_cast<float4*>(&As[s][r0][q0])      = pa0;
    *reinterpret_cast<float4*>(&As[s][r0 + 64][q0]) = pa1;
    *reinterpret_cast<float4*>(&Bs[s][r0][q0])      = pb0;
    *reinterpret_cast<float4*>(&Bs[s][r0 + 64][q0]) = pb1;
  };
  ldreg(0); stor(0);
  for (int i = 0; i < nk; ++i) {
    const bool next = (i + 1) < nk;
    if (next) ldreg((i + 1) * 32);   // global loads in flight during compute
    __syncthreads();
    compute(i % 3);
    if (next) stor((i + 1) % 3);
  }
#endif

  // Epilogue: bias (+ exact GELU); normal or transposed store.
  // C/D layout: VGPR r of lane -> row m = r + 8*half, col n = l16.
#pragma unroll
  for (int i = 0; i < 2; ++i) {
#pragma unroll
    for (int j = 0; j < 4; ++j) {
      const long long n  = n0 + wn * 64 + j * 16 + l16;
      const float bv     = bias ? bias[n] : 0.0f;
      const long long mb = m0 + wm * 32 + i * 16 + 8 * half;
#pragma unroll
      for (int r = 0; r < 8; ++r) {
        float vv = acc[i][j][r] + bv;
        if (DO_GELU) vv = 0.5f * vv * (1.0f + erff(vv * 0.70710678118654752f));
        const long long idx = OUT_TRANS ? n * (long long)ldc + (mb + r)
                                        : (mb + r) * (long long)ldc + n;
        if (OUT_F32) {
          ((float*)Cout + (long long)blockIdx.z * sC)[idx] = vv;
        } else {
          ((uint16_t*)Cout + (long long)blockIdx.z * sC)[idx] = f2bf(vv);
        }
      }
    }
  }
}

// Row softmax over 2048 f32 columns -> bf16 probabilities. One block per row.
__global__ __launch_bounds__(256) void softmax_row_2048(
    const float* __restrict__ S, uint16_t* __restrict__ P) {
  const long long row = blockIdx.x;
  const int t = threadIdx.x;
  const float* sr = S + row * 2048;
  float x[8];
  float mx = -3.4e38f;
#pragma unroll
  for (int i = 0; i < 8; ++i) { x[i] = sr[t + 256 * i]; mx = fmaxf(mx, x[i]); }
  __shared__ float red[256];
  red[t] = mx; __syncthreads();
  for (int s = 128; s > 0; s >>= 1) {
    if (t < s) red[t] = fmaxf(red[t], red[t + s]);
    __syncthreads();
  }
  mx = red[0]; __syncthreads();
  float sum = 0.f;
#pragma unroll
  for (int i = 0; i < 8; ++i) { x[i] = expf(x[i] - mx); sum += x[i]; }
  red[t] = sum; __syncthreads();
  for (int s = 128; s > 0; s >>= 1) {
    if (t < s) red[t] += red[t + s];
    __syncthreads();
  }
  const float inv = 1.0f / red[0];
  uint16_t* pr = P + row * 2048;
#pragma unroll
  for (int i = 0; i < 8; ++i) pr[t + 256 * i] = f2bf(x[i] * inv);
}

extern "C" void kernel_launch(void* const* d_in, const int* in_sizes, int n_in,
                              void* d_out, int out_size, void* d_ws, size_t ws_size,
                              hipStream_t stream) {
  (void)in_sizes; (void)n_in; (void)out_size; (void)ws_size;
  const float* q_in  = (const float*)d_in[0];
  const float* kv_in = (const float*)d_in[1];
  const float* W_kv  = (const float*)d_in[2];
  const float* b_kv  = (const float*)d_in[3];
  const float* W_q   = (const float*)d_in[4];
  const float* b_q   = (const float*)d_in[5];
  const float* W1    = (const float*)d_in[6];
  const float* b1    = (const float*)d_in[7];
  const float* W2    = (const float*)d_in[8];
  const float* b2    = (const float*)d_in[9];
  float* out = (float*)d_out;

  char* ws = (char*)d_ws;
  const size_t MiB = 1ull << 20;
  uint16_t* qact   = (uint16_t*)(ws +   0 * MiB);  //  32 MiB  _q bf16
  uint16_t* kvact  = (uint16_t*)(ws +  32 * MiB);  //  32 MiB  _kv bf16
  uint16_t* wkvT   = (uint16_t*)(ws +  64 * MiB);  //   4 MiB  W_kv^T [2048][1024]
  uint16_t* wqT    = (uint16_t*)(ws +  68 * MiB);  //   2 MiB  W_q^T  [1024][1024]
  uint16_t* w1T    = (uint16_t*)(ws +  70 * MiB);  //   8 MiB  W1^T   [4096][1024]
  uint16_t* w2T    = (uint16_t*)(ws +  78 * MiB);  //   8 MiB  W2^T   [1024][4096]
  uint16_t* qproj  = (uint16_t*)(ws +  86 * MiB);  //  32 MiB  q   [16384][1024]
  uint16_t* kbuf   = (uint16_t*)(ws + 118 * MiB);  //  32 MiB  k   [16384][1024]
  uint16_t* vT     = (uint16_t*)(ws + 150 * MiB);  //  32 MiB  v^T [8][1024][2048]
  uint16_t* outb   = (uint16_t*)(ws + 182 * MiB);  //  32 MiB  attn out
  uint16_t* probs  = (uint16_t*)(ws + 214 * MiB);  //  64 MiB
  float*    scores = (float*)   (ws + 278 * MiB);  // 128 MiB  f32
  uint16_t* hidden = (uint16_t*)(ws + 278 * MiB);  // 128 MiB  (reuses scores)

  dim3 blk(256);
  auto cvt = [&](const float* src, uint16_t* dst, long long n) {
    cvt_f32_bf16<<<dim3((unsigned)((n + 255) / 256)), blk, 0, stream>>>(src, dst, n);
  };
  cvt(q_in,  qact,  BS * Hc);
  cvt(kv_in, kvact, BS * Hc);
  cvt_transpose_f32_bf16<<<dim3(2048 / 32, 1024 / 32), blk, 0, stream>>>(W_kv, wkvT, 1024, 2048);
  cvt_transpose_f32_bf16<<<dim3(1024 / 32, 1024 / 32), blk, 0, stream>>>(W_q,  wqT,  1024, 1024);
  cvt_transpose_f32_bf16<<<dim3(4096 / 32, 1024 / 32), blk, 0, stream>>>(W1,   w1T,  1024, 4096);
  cvt_transpose_f32_bf16<<<dim3(1024 / 32, 4096 / 32), blk, 0, stream>>>(W2,   w2T,  4096, 1024);

  // k = _kv @ Wk + bk  -> bf16 [16384 x 1024]
  gemm_nt_bf16_wmma<false, false, false><<<dim3(8, 128, 1), blk, 0, stream>>>(
      kvact, wkvT, b_kv, kbuf, 1024, 1024, 1024, 1024, 0, 0, 0);
  // v^T[b] = (_kv[b] @ Wv + bv)^T -> bf16 [1024 x 2048] per batch
  gemm_nt_bf16_wmma<false, false, true><<<dim3(8, 16, 8), blk, 0, stream>>>(
      kvact, wkvT + 1024 * 1024, b_kv + 1024, vT, 1024, 1024, 1024, 2048,
      2048LL * 1024, 0, 1024LL * 2048);
  // q = _q @ Wq + bq -> bf16 [16384 x 1024]
  gemm_nt_bf16_wmma<false, false, false><<<dim3(8, 128, 1), blk, 0, stream>>>(
      qact, wqT, b_q, qproj, 1024, 1024, 1024, 1024, 0, 0, 0);
  // scores[b] = q[b] @ k[b]^T -> f32 [2048 x 2048] per batch
  gemm_nt_bf16_wmma<false, true, false><<<dim3(16, 16, 8), blk, 0, stream>>>(
      qproj, kbuf, nullptr, scores, 1024, 1024, 1024, 2048,
      2048LL * 1024, 2048LL * 1024, 2048LL * 2048);
  // probs = softmax(scores) -> bf16
  softmax_row_2048<<<dim3(16384), blk, 0, stream>>>(scores, probs);
  // out[b] = probs[b] @ (v^T[b])^T -> bf16 [2048 x 1024] per batch
  gemm_nt_bf16_wmma<false, false, false><<<dim3(8, 16, 8), blk, 0, stream>>>(
      probs, vT, nullptr, outb, 2048, 2048, 2048, 1024,
      2048LL * 2048, 1024LL * 2048, 2048LL * 1024);
  // hidden = gelu(out @ W1 + b1) -> bf16 [16384 x 4096]
  gemm_nt_bf16_wmma<true, false, false><<<dim3(32, 128, 1), blk, 0, stream>>>(
      outb, w1T, b1, hidden, 1024, 1024, 1024, 4096, 0, 0, 0);
  // y = hidden @ W2 + b2 -> f32 d_out [16384 x 1024]
  gemm_nt_bf16_wmma<false, true, false><<<dim3(8, 128, 1), blk, 0, stream>>>(
      hidden, w2T, b2, out, 4096, 4096, 4096, 1024, 0, 0, 0);
}
